// Encoder_78048145702983
// MI455X (gfx1250) — compile-verified, matrix operand order checked by reading, not run on hardware
//
#include <hip/hip_runtime.h>
#include <hip/hip_bf16.h>
#include <hip/hip_fp16.h>

// Problem dims
#define TT 512
#define BB 32
#define EE 512
#define HH 512
#define NG 2048        // 4*H gate width
#define SL (BB*HH)     // one h slab (elements)

typedef _Float16 half_t;
typedef __attribute__((ext_vector_type(16))) _Float16 v16h;
typedef __attribute__((ext_vector_type(8)))  float    v8f;
typedef __attribute__((ext_vector_type(4)))  unsigned int v4u;
typedef __attribute__((ext_vector_type(8)))  int      v8i;
typedef __attribute__((ext_vector_type(4)))  int      v4i;

union F16x8 { uint4 u; half_t h[8]; };

// Load a 16x32 (f16) WMMA fragment. base is row-major with `stride` halves per row.
// Lane l<16 holds row tile0+l, K halves {k0..k0+7, k0+16..k0+23}; lanes 16..31 the +8 halves.
__device__ __forceinline__ v16h load_frag(const half_t* base, int stride, int tile0, int k0, int lane) {
  int r  = tile0 + (lane & 15);
  int kk = k0 + ((lane >> 4) << 3);
  F16x8 lo, hi;
  lo.u = *(const uint4*)(base + (size_t)r * stride + kk);
  hi.u = *(const uint4*)(base + (size_t)r * stride + kk + 16);
  v16h v;
#pragma unroll
  for (int i = 0; i < 8; ++i) { v[i] = lo.h[i]; v[i + 8] = hi.h[i]; }
  return v;
}

__device__ __forceinline__ v8f wmma_f16(v16h a, v16h b, v8f c) {
  return __builtin_amdgcn_wmma_f32_16x16x32_f16(false, a, false, b, (short)0, c, false, false);
}

__device__ __forceinline__ float sigmoidf_(float x) { return 1.0f / (1.0f + __expf(-x)); }

// ---------------------------------------------------------------------------
// Stage one contiguous 32x512 f16 slab (32 KB) from global into LDS with a
// 16B-per-1024B-row pad (row pitch 520 halves) to avoid LDS bank camping.
// Preferred path: Tensor Data Mover (async DMA, TENSORcnt). The D# descriptor
// uses data_size=8B, a 1-D tile of 4096 qwords, pad_interval=7 (every 256
// DWORDs = 1024B) and pad_amount=3 (4 DWORDs = 16B) so the DMA engine lays the
// data out at the padded pitch for free. One wave issues; all waves barrier.
// ---------------------------------------------------------------------------
__device__ __forceinline__ void stage_slab_lds(const half_t* __restrict__ gsrc,
                                               half_t* lds, int tid) {
#if __has_builtin(__builtin_amdgcn_tensor_load_to_lds)
  if (tid < 32) {
    unsigned long long ga = (unsigned long long)(uintptr_t)gsrc;
    unsigned ldsoff = (unsigned)(uintptr_t)lds;   // low 32 bits of generic = LDS offset
    v4u g0;
    g0[0] = 1u;                                            // count=1, load descriptor
    g0[1] = ldsoff;                                        // lds_addr
    g0[2] = (unsigned)(ga & 0xFFFFFFFFu);                  // global_addr[31:0]
    g0[3] = (unsigned)((ga >> 32) & 0x1FFFFFFu) | (2u << 30); // global_addr[56:32], type=2
    v8i g1;
    g1[0] = (3 << 16)      // data_size = 8B
          | (1 << 20)      // pad_enable
          | (7 << 22)      // pad_interval: every 256 DWORDs (1024B row)
          | (3 << 25);     // pad_amount: 4 DWORDs (16B)
    g1[1] = (int)(4096u << 16);   // tensor_dim0[15:0] = 4096 qwords (in [31:16])
    g1[2] = (1 << 16);            // tensor_dim0 hi = 0, tensor_dim1 lo = 1
    g1[3] = (int)(4096u << 16);   // tensor_dim1 hi = 0, tile_dim0 = 4096
    g1[4] = 1;                    // tile_dim1 = 1, tile_dim2 = 0
    g1[5] = 4096;                 // tensor_dim0_stride lo
    g1[6] = 0;                    // tensor_dim0_stride hi, tensor_dim1_stride lo
    g1[7] = 0;
    v4i gz = {0, 0, 0, 0};
#if defined(__clang_major__) && __clang_major__ >= 23
    v8i gz8 = {0, 0, 0, 0, 0, 0, 0, 0};
    __builtin_amdgcn_tensor_load_to_lds(g0, g1, gz, gz, gz8, 0);
#else
    __builtin_amdgcn_tensor_load_to_lds(g0, g1, gz, gz, 0);
#endif
    __builtin_amdgcn_s_wait_tensorcnt(0);
  }
  __syncthreads();
#else
  for (int i = tid; i < 32 * 64; i += 256) {    // 64 uint4 (=512 halves) per row
    int row = i >> 6, c8 = i & 63;
    *(uint4*)(lds + row * 520 + c8 * 8) = *(const uint4*)(gsrc + row * 512 + c8 * 8);
  }
  __syncthreads();
#endif
}

// ---------------- embedding gather + f32->f16, time-major [T][B][E] ----------------
__global__ void embed_kernel(const int* __restrict__ tokens, const float* __restrict__ emb,
                             half_t* __restrict__ X16) {
  int idx = blockIdx.x * blockDim.x + threadIdx.x;   // over T*B*(E/4)
  int e4 = idx & 127;          // E/4 = 128
  int tb = idx >> 7;
  int b  = tb & 31;
  int t  = tb >> 5;
  int tok = tokens[b * TT + t];
  float4 v = *(const float4*)(emb + (size_t)tok * EE + e4 * 4);
  half_t* dst = X16 + ((size_t)(t * BB + b)) * EE + e4 * 4;
  dst[0] = (half_t)v.x; dst[1] = (half_t)v.y; dst[2] = (half_t)v.z; dst[3] = (half_t)v.w;
}

// ---------------- weight f32 [K=512][N=2048] -> f16 transposed [N][K] ----------------
__global__ void transpose_w_kernel(const float* __restrict__ W, half_t* __restrict__ WT) {
  int idx = blockIdx.x * blockDim.x + threadIdx.x;   // over 2048*512
  int n = idx >> 9;
  int k = idx & 511;
  WT[idx] = (half_t)W[(size_t)k * NG + n];
}

// ---------------- big precompute GEMM: Zx[t] = X[t] @ Wx0 (both directions) ----------------
// grid: (T, 2).  Block stages the 32x512 X slab in LDS via TDM, 8 waves sweep 2x128 tiles.
__global__ void __launch_bounds__(256) zx_gemm_kernel(const half_t* __restrict__ X16,
                                                      const half_t* __restrict__ WxT_f,
                                                      const half_t* __restrict__ WxT_b,
                                                      half_t* __restrict__ ZxF,
                                                      half_t* __restrict__ ZxB) {
  int t = blockIdx.x;
  const half_t* WT = blockIdx.y ? WxT_b : WxT_f;
  half_t* Z = (blockIdx.y ? ZxB : ZxF) + (size_t)t * (BB * NG);

  __shared__ __align__(16) half_t a_lds[32 * 520];   // padded rows: 520 halves
  stage_slab_lds(X16 + (size_t)t * (BB * EE), a_lds, threadIdx.x);

  int wave = threadIdx.x >> 5, lane = threadIdx.x & 31;
  for (int tile = wave; tile < 256; tile += 8) {     // 2 m-tiles x 128 n-tiles
    int m0 = (tile & 1) * 16;
    int n0 = (tile >> 1) * 16;
    v8f acc = {};
    for (int k0 = 0; k0 < 512; k0 += 32) {
      v16h a = load_frag(a_lds, 520, m0, k0, lane);
      v16h b = load_frag(WT, 512, n0, k0, lane);
      acc = wmma_f16(a, b, acc);
    }
    int col  = n0 + (lane & 15);
    int rowB = m0 + ((lane >> 4) << 3);
#pragma unroll
    for (int r = 0; r < 8; ++r) Z[(size_t)(rowB + r) * NG + col] = (half_t)acc[r];
  }
}

// ---------------- one LSTM time step (both directions via blockIdx.y) ----------------
struct StepArgs {
  const half_t* xIn0;  const half_t* xIn1;    // layer input f16 [32][512] (null => use zPre)
  const half_t* WxT0;  const half_t* WxT1;    // f16 [2048][512] (null for layer 0)
  const half_t* hPrev0; const half_t* hPrev1; // f16 [32][512]
  const half_t* WhT0;  const half_t* WhT1;    // f16 [2048][512]
  const half_t* zPre0; const half_t* zPre1;   // f16 [32][2048] precomputed x@Wx (layer 0)
  const float*  bias0; const float*  bias1;   // f32 [2048]
  float* c0; float* c1;                       // f32 [32][512] in-place cell state
  half_t* hOut0; half_t* hOut1;               // f16 [32][512]
};

__global__ void __launch_bounds__(256) lstm_step_kernel(StepArgs A) {
  int d = blockIdx.y;
  const half_t* xIn   = d ? A.xIn1  : A.xIn0;
  const half_t* WxT   = d ? A.WxT1  : A.WxT0;
  const half_t* hPrev = d ? A.hPrev1 : A.hPrev0;
  const half_t* WhT   = d ? A.WhT1  : A.WhT0;
  const half_t* zPre  = d ? A.zPre1 : A.zPre0;
  const float*  bias  = d ? A.bias1 : A.bias0;
  float*  cState      = d ? A.c1    : A.c0;
  half_t* hOut        = d ? A.hOut1 : A.hOut0;

  __shared__ __align__(16) half_t h_lds[32 * 520];
  stage_slab_lds(hPrev, h_lds, threadIdx.x);

  int wave = threadIdx.x >> 5, lane = threadIdx.x & 31;
  int tile = blockIdx.x * 8 + wave;     // 64 tiles: 2 m x 32 n (h columns)
  int m0 = (tile & 1) * 16;
  int n0 = (tile >> 1) * 16;

  v8f acc[4] = {};                      // i, f, g, o accumulators
  for (int k0 = 0; k0 < 512; k0 += 32) {
    // keep the next weight chunk warm in L2/L0
    __builtin_prefetch(WhT + (size_t)n0 * 512 + k0 + 32, 0, 1);
    v16h ah = load_frag(h_lds, 520, m0, k0, lane);
#pragma unroll
    for (int g = 0; g < 4; ++g) {
      v16h bw = load_frag(WhT, 512, g * HH + n0, k0, lane);
      acc[g] = wmma_f16(ah, bw, acc[g]);
    }
    if (xIn) {                          // layer >= 1: fused x @ Wx
      v16h ax = load_frag(xIn, 512, m0, k0, lane);
#pragma unroll
      for (int g = 0; g < 4; ++g) {
        v16h bw = load_frag(WxT, 512, g * HH + n0, k0, lane);
        acc[g] = wmma_f16(ax, bw, acc[g]);
      }
    }
  }

  int col  = n0 + (lane & 15);              // h column 0..511
  int rowB = m0 + ((lane >> 4) << 3);
  float bi = bias[col], bf = bias[HH + col], bg = bias[2 * HH + col], bo = bias[3 * HH + col];
#pragma unroll
  for (int r = 0; r < 8; ++r) {
    int row = rowB + r;
    float zi = acc[0][r] + bi;
    float zf = acc[1][r] + bf;
    float zg = acc[2][r] + bg;
    float zo = acc[3][r] + bo;
    if (zPre) {
      const half_t* zp = zPre + (size_t)row * NG;
      zi += (float)zp[col];            zf += (float)zp[HH + col];
      zg += (float)zp[2 * HH + col];   zo += (float)zp[3 * HH + col];
    }
    float iG = sigmoidf_(zi), fG = sigmoidf_(zf), gG = tanhf(zg), oG = sigmoidf_(zo);
    float cP = cState[(size_t)row * HH + col];
    float cN = fG * cP + iG * gG;
    cState[(size_t)row * HH + col] = cN;
    hOut[(size_t)row * HH + col] = (half_t)(oG * tanhf(cN));
  }
}

// ---------------- final combine: out = 0.5*(h_f1 + h_b1), plus last-step state ----------------
__global__ void combine_kernel(const half_t* __restrict__ hF1, const half_t* __restrict__ hB1,
                               float* __restrict__ out) {
  int idx = blockIdx.x * blockDim.x + threadIdx.x;  // over B*T*E, flat [B][T][E]
  int e  = idx & 511;
  int bt = idx >> 9;
  int t  = bt & 511;
  int b  = bt >> 9;
  float yf = (float)hF1[((size_t)(t + 1) * BB + b) * HH + e];          // fwd step t
  float yb = (float)hB1[((size_t)(TT - t) * BB + b) * HH + e];         // bwd step T-1-t (+1 shift)
  float v = 0.5f * (yf + yb);
  out[idx] = v;
  if (t == TT - 1) out[(size_t)BB * TT * EE + (size_t)b * EE + e] = v; // state
}

extern "C" void kernel_launch(void* const* d_in, const int* in_sizes, int n_in,
                              void* d_out, int out_size, void* d_ws, size_t ws_size,
                              hipStream_t stream) {
  const int*   tokens = (const int*)d_in[0];
  const float* emb    = (const float*)d_in[1];
  // dict order: Wx_f0, Wh_f0, Wx_f1, Wh_f1, Wx_b0, Wh_b0, Wx_b1, Wh_b1, b_f0, b_f1, b_b0, b_b1
  const float* Wsrc[8];
  for (int i = 0; i < 8; ++i) Wsrc[i] = (const float*)d_in[2 + i];
  const float* b_f0 = (const float*)d_in[10];
  const float* b_f1 = (const float*)d_in[11];
  const float* b_b0 = (const float*)d_in[12];
  const float* b_b1 = (const float*)d_in[13];

  char* ws = (char*)d_ws;
  size_t off = 0;
  auto alloc = [&](size_t bytes) -> void* {
    void* p = ws + off;
    off += (bytes + 255) & ~(size_t)255;
    return p;
  };

  half_t* X16 = (half_t*)alloc((size_t)TT * BB * EE * 2);
  half_t* WT[8];                                            // transposed f16 weights [N][K]
  for (int i = 0; i < 8; ++i) WT[i] = (half_t*)alloc((size_t)NG * 512 * 2);
  half_t* ZxF = (half_t*)alloc((size_t)TT * BB * NG * 2);
  half_t* ZxB = (half_t*)alloc((size_t)TT * BB * NG * 2);
  half_t* hSeq[4];                                          // F0, F1, B0, B1 : [(T+1)][32][512]
  for (int i = 0; i < 4; ++i) hSeq[i] = (half_t*)alloc((size_t)(TT + 1) * SL * 2);
  float* cBuf[4];
  for (int i = 0; i < 4; ++i) cBuf[i] = (float*)alloc((size_t)SL * 4);

  // zero-init h(-1) slabs and cell states (deterministic per call)
  for (int i = 0; i < 4; ++i) {
    hipMemsetAsync(hSeq[i], 0, (size_t)SL * 2, stream);
    hipMemsetAsync(cBuf[i], 0, (size_t)SL * 4, stream);
  }

  embed_kernel<<<(TT * BB * (EE / 4)) / 256, 256, 0, stream>>>(tokens, emb, X16);
  for (int i = 0; i < 8; ++i)
    transpose_w_kernel<<<(NG * 512) / 256, 256, 0, stream>>>(Wsrc[i], WT[i]);

  // Hoisted layer-0 input GEMMs (fwd uses Wx_f0 = WT[0], bwd uses Wx_b0 = WT[4])
  zx_gemm_kernel<<<dim3(TT, 2), 256, 0, stream>>>(X16, WT[0], WT[4], ZxF, ZxB);

  half_t *hF0 = hSeq[0], *hF1 = hSeq[1], *hB0 = hSeq[2], *hB1 = hSeq[3];

  for (int s = 0; s < TT; ++s) {
    // ---- layer 0 (uses precomputed x@Wx; bwd indexes reversed time) ----
    StepArgs a0{};
    a0.xIn0 = nullptr;                 a0.xIn1 = nullptr;
    a0.WxT0 = nullptr;                 a0.WxT1 = nullptr;
    a0.hPrev0 = hF0 + (size_t)s * SL;  a0.hPrev1 = hB0 + (size_t)s * SL;
    a0.WhT0 = WT[1];                   a0.WhT1 = WT[5];
    a0.zPre0 = ZxF + (size_t)s * BB * NG;
    a0.zPre1 = ZxB + (size_t)(TT - 1 - s) * BB * NG;
    a0.bias0 = b_f0;                   a0.bias1 = b_b0;
    a0.c0 = cBuf[0];                   a0.c1 = cBuf[2];
    a0.hOut0 = hF0 + (size_t)(s + 1) * SL;
    a0.hOut1 = hB0 + (size_t)(s + 1) * SL;
    lstm_step_kernel<<<dim3(8, 2), 256, 0, stream>>>(a0);

    // ---- layer 1 (fused x@Wx + h@Wh) ----
    StepArgs a1{};
    a1.xIn0 = hF0 + (size_t)(s + 1) * SL;  a1.xIn1 = hB0 + (size_t)(s + 1) * SL;
    a1.WxT0 = WT[2];                       a1.WxT1 = WT[6];
    a1.hPrev0 = hF1 + (size_t)s * SL;      a1.hPrev1 = hB1 + (size_t)s * SL;
    a1.WhT0 = WT[3];                       a1.WhT1 = WT[7];
    a1.zPre0 = nullptr;                    a1.zPre1 = nullptr;
    a1.bias0 = b_f1;                       a1.bias1 = b_b1;
    a1.c0 = cBuf[1];                       a1.c1 = cBuf[3];
    a1.hOut0 = hF1 + (size_t)(s + 1) * SL;
    a1.hOut1 = hB1 + (size_t)(s + 1) * SL;
    lstm_step_kernel<<<dim3(8, 2), 256, 0, stream>>>(a1);
  }

  combine_kernel<<<((size_t)BB * TT * EE) / 256, 256, 0, stream>>>(hF1, hB1, (float*)d_out);
}
